// imp_84610855731247
// MI455X (gfx1250) — matrix operand order repaired and checked by reading, not compiled, not run on hardware
//
#include <hip/hip_runtime.h>
#include <math.h>

// ---------------------------------------------------------------------------
// Problem dims (fixed by the reference): B=16, N=64, P=128, EF=512, H=128,
// CH=512. Edge rows ME = B*N*N = 65536. Node rows = 1024. Pair rows = 2048.
// ---------------------------------------------------------------------------

typedef __attribute__((ext_vector_type(16))) __bf16 v16bf;
typedef __attribute__((ext_vector_type(8)))  float  v8f;

union Frag16 { v16bf v; unsigned int u[8]; };
union BF4    { __bf16 h[4]; uint2 u2; };
union BF8    { __bf16 h[8]; uint4 u4; };

__device__ __forceinline__ float sigm(float x) { return 1.0f / (1.0f + __expf(-x)); }

// ---------------------------------------------------------------------------
// One-shot weight prep kernels (run once; sizes are tiny).
// ---------------------------------------------------------------------------
// y[i] = bf16(x[i]); n multiple of 4.
__global__ __launch_bounds__(256) void convert_bf16(
    const float* __restrict__ x, __bf16* __restrict__ y, int n) {
  const int i = (blockIdx.x * 256 + threadIdx.x) * 4;
  if (i >= n) return;
  const float4 v = *(const float4*)(x + i);
  BF4 o;
  o.h[0] = (__bf16)v.x; o.h[1] = (__bf16)v.y;
  o.h[2] = (__bf16)v.z; o.h[3] = (__bf16)v.w;
  *(uint2*)(y + i) = o.u2;
}

// Wt[n][k] = bf16(B[k][n]); B is [K,N] row-major. Writes coalesced over k.
__global__ __launch_bounds__(256) void transpose_to_bf16(
    const float* __restrict__ B, __bf16* __restrict__ Wt, int K, int N) {
  const int idx = blockIdx.x * 256 + threadIdx.x;  // idx = n*K + k
  if (idx >= N * K) return;
  const int n = idx / K, k = idx % K;
  Wt[idx] = (__bf16)B[(size_t)k * N + n];
}

// ---------------------------------------------------------------------------
// bf16 WMMA GEMM:  C[M,N] = act(A[M,K] @ W + bias), W pre-transposed bf16
// Wt[N][K].  Block = 256 threads (8 waves), tile 64(M) x 128(N), BK = 32.
// Wave w: wm = w&3 (16-row tile), wn = w>>2 (64-col half) -> 4 accum tiles.
// All 4 B fragments are preloaded before the WMMA burst to avoid
// WMMA->VALU WAR hazard NOPs on fragment register reuse.
// ---------------------------------------------------------------------------
template <bool A_IS_F32, bool RELU>
__global__ __launch_bounds__(256) void gemm_bias_wmma(
    const void* __restrict__ Ap, const __bf16* __restrict__ Wt,
    const float* __restrict__ bias, __bf16* __restrict__ C,
    int M, int N, int K) {
  __shared__ __align__(16) __bf16 Ash[64][36];   // 32 + 4 pad (72B rows)
  __shared__ __align__(16) __bf16 Bsh[128][36];  // Bsh[n][k]

  const int tid = threadIdx.x;
  const int w   = tid >> 5;
  const int l   = tid & 31;
  const int hi  = l >> 4;
  const int l16 = l & 15;
  const int wm  = w & 3;
  const int wn  = w >> 2;
  const int m0  = blockIdx.x * 64;
  const int n0  = blockIdx.y * 128;

  const float*  Af = (const float*)Ap;
  const __bf16* Ab = (const __bf16*)Ap;
  const char*   Ac = (const char*)Ap;
  const size_t  esz = A_IS_F32 ? 4 : 2;

  // A staging: 64x32 = 2048 elems, 8 per thread
  const int ar  = tid >> 2;          // row 0..63
  const int ac0 = (tid & 3) * 8;     // col 0,8,16,24
  // B staging: 128x32 = 4096 elems, 16 per thread as 2 x uint4 (8 bf16 each)
  const int bn0 = tid >> 2;
  const int bc0 = (tid & 3) * 8;

  v8f zero = {};
  v8f acc[4];
#pragma unroll
  for (int t = 0; t < 4; ++t) acc[t] = zero;

  for (int k0 = 0; k0 < K; k0 += 32) {
    // ---- Stage A tile (128-bit global loads, 64-bit LDS stores) ----
    if (A_IS_F32) {
      const float4* p =
          (const float4*)(Af + (size_t)(m0 + ar) * K + (k0 + ac0));
      const float4 v0 = p[0], v1 = p[1];
      BF4 x0, x1;
      x0.h[0] = (__bf16)v0.x; x0.h[1] = (__bf16)v0.y;
      x0.h[2] = (__bf16)v0.z; x0.h[3] = (__bf16)v0.w;
      x1.h[0] = (__bf16)v1.x; x1.h[1] = (__bf16)v1.y;
      x1.h[2] = (__bf16)v1.z; x1.h[3] = (__bf16)v1.w;
      *(uint2*)&Ash[ar][ac0]     = x0.u2;
      *(uint2*)&Ash[ar][ac0 + 4] = x1.u2;
    } else {
      BF8 x;
      x.u4 = *(const uint4*)(Ab + (size_t)(m0 + ar) * K + (k0 + ac0));
      *(uint2*)&Ash[ar][ac0]     = make_uint2(x.u4.x, x.u4.y);
      *(uint2*)&Ash[ar][ac0 + 4] = make_uint2(x.u4.z, x.u4.w);
    }
    // ---- Stage B tile: straight bf16 copy from pre-transposed weights ----
#pragma unroll
    for (int i = 0; i < 2; ++i) {
      const int n = bn0 + i * 64;  // 0..127
      BF8 x;
      x.u4 = *(const uint4*)(Wt + (size_t)(n0 + n) * K + (k0 + bc0));
      *(uint2*)&Bsh[n][bc0]     = make_uint2(x.u4.x, x.u4.y);
      *(uint2*)&Bsh[n][bc0 + 4] = make_uint2(x.u4.z, x.u4.w);
    }
    if (k0 + 32 < K)
      __builtin_prefetch(Ac + ((size_t)(m0 + ar) * K + (k0 + 32)) * esz, 0, 0);
    __syncthreads();

    // A fragment: 16-bit A layout -> k pairs {0,2,4,6}/{16,18,20,22} (+8 if hi)
    Frag16 fa;
#pragma unroll
    for (int v = 0; v < 8; ++v) {
      const int kk = ((v & 4) ? (16 + ((v & 3) << 1)) : (v << 1)) + (hi << 3);
      fa.u[v] = *(const unsigned int*)&Ash[wm * 16 + l16][kk];
    }
    // Preload all B fragments, then burst the WMMAs (no WAR reuse stalls)
    Frag16 fb[4];
#pragma unroll
    for (int t = 0; t < 4; ++t) {
      const int nl = wn * 64 + t * 16 + l16;
#pragma unroll
      for (int v = 0; v < 8; ++v) {
        const int kk = ((v & 4) ? (16 + ((v & 3) << 1)) : (v << 1)) + (hi << 3);  // B layout: k pairs {0,2,4,6}/{16,18,20,22} (+8 hi)
        fb[t].u[v] = *(const unsigned int*)&Bsh[nl][kk];
      }
    }
#pragma unroll
    for (int t = 0; t < 4; ++t)
      acc[t] = __builtin_amdgcn_wmma_f32_16x16x32_bf16(
          false, fa.v, false, fb[t].v, (short)0, acc[t], false, false);
    __syncthreads();
  }

  // C/D layout: VGPR r -> M = r (lanes 0-15) / r+8 (lanes 16-31), N = lane&15
#pragma unroll
  for (int t = 0; t < 4; ++t) {
    const int n = n0 + wn * 64 + t * 16 + l16;
    const float bv = bias[n];
#pragma unroll
    for (int r = 0; r < 8; ++r) {
      const int m = m0 + wm * 16 + r + (hi << 3);
      float val = acc[t][r] + bv;
      if (RELU) val = fmaxf(val, 0.0f);
      C[(size_t)m * N + n] = (__bf16)val;
    }
  }
}

// ---------------------------------------------------------------------------
// Fused GRU step: G = X[M,128] @ wih^T (-> [M,384]) ; gates ; H = (1-z)*n.
// Block = 256 threads, 64 rows per block. The whole A block (64x128) and the
// whole bf16 weight matrix (384x128, = wih row-major) are loaded into LDS
// ONCE, then 4 K-steps of WMMA run barrier-free from LDS.
// Wave w: wm = w&3 (16 rows), ng = w>>2 (0..1, 192 cols = 12 tiles).
// All 12 B fragments per K-step preloaded before the WMMA burst.
// ---------------------------------------------------------------------------
__global__ __launch_bounds__(256) void gru_wmma(
    const __bf16* __restrict__ X, const __bf16* __restrict__ wihb,
    const float* __restrict__ bih, const float* __restrict__ bhh,
    float* __restrict__ Hout) {
  __shared__ __align__(16) union SM {
    struct { __bf16 A[64][132]; __bf16 Bt[384][132]; } s;  // full-K tiles
    float G[64][388];
  } sm;

  const int tid = threadIdx.x;
  const int w   = tid >> 5;
  const int l   = tid & 31;
  const int hi  = l >> 4;
  const int l16 = l & 15;
  const int wm  = w & 3;        // 16-row tile of 64
  const int ng  = w >> 2;       // 0..1 -> 192 cols (12 tiles)
  const int row0 = blockIdx.x * 64;

  // Staging chunks of 8 bf16 (uint4): coords from chunk index q.
  const int qr = tid >> 4;          // +16 rows per iteration
  const int qc = (tid & 15) * 8;    // 0..120

  v8f zero = {};
  v8f acc[12];
#pragma unroll
  for (int t = 0; t < 12; ++t) acc[t] = zero;

  // ---- Load A (64x128) once: 1024 chunks, 4 per thread ----
#pragma unroll
  for (int i = 0; i < 4; ++i) {
    const int r = qr + i * 16;
    BF8 x;
    x.u4 = *(const uint4*)(X + (size_t)(row0 + r) * 128 + qc);
    *(uint2*)&sm.s.A[r][qc]     = make_uint2(x.u4.x, x.u4.y);
    *(uint2*)&sm.s.A[r][qc + 4] = make_uint2(x.u4.z, x.u4.w);
  }
  // ---- Load Wt (384x128 bf16) once: 6144 chunks, 24 per thread ----
#pragma unroll
  for (int i = 0; i < 24; ++i) {
    const int r = qr + i * 16;
    BF8 x;
    x.u4 = *(const uint4*)(wihb + (size_t)r * 128 + qc);
    *(uint2*)&sm.s.Bt[r][qc]     = make_uint2(x.u4.x, x.u4.y);
    *(uint2*)&sm.s.Bt[r][qc + 4] = make_uint2(x.u4.z, x.u4.w);
  }
  __syncthreads();

  // ---- 4 K-steps, all LDS-resident, no barriers ----
  for (int k0 = 0; k0 < 128; k0 += 32) {
    Frag16 fa;
#pragma unroll
    for (int v = 0; v < 8; ++v) {
      const int kk =
          k0 + ((v & 4) ? (16 + ((v & 3) << 1)) : (v << 1)) + (hi << 3);
      fa.u[v] = *(const unsigned int*)&sm.s.A[wm * 16 + l16][kk];
    }
    // Preload all 12 B fragments into distinct registers
    Frag16 fb[12];
#pragma unroll
    for (int t = 0; t < 12; ++t) {
      const int nl = ng * 192 + t * 16 + l16;
#pragma unroll
      for (int v = 0; v < 8; ++v) {
        const int kk =
            k0 + ((v & 4) ? (16 + ((v & 3) << 1)) : (v << 1)) + (hi << 3);
        fb[t].u[v] = *(const unsigned int*)&sm.s.Bt[nl][kk];
      }
    }
    // Burst 12 WMMAs back-to-back (distinct acc, distinct B regs)
#pragma unroll
    for (int t = 0; t < 12; ++t)
      acc[t] = __builtin_amdgcn_wmma_f32_16x16x32_bf16(
          false, fa.v, false, fb[t].v, (short)0, acc[t], false, false);
  }
  __syncthreads();  // all waves done reading s.A/s.Bt before G overwrites

  // Spill G (64x384 f32) into LDS (union reuse)
#pragma unroll
  for (int t = 0; t < 12; ++t) {
#pragma unroll
    for (int r = 0; r < 8; ++r) {
      const int ml = wm * 16 + r + (hi << 3);
      const int nl = ng * 192 + t * 16 + l16;
      sm.G[ml][nl] = acc[t][r];
    }
  }
  __syncthreads();

  // Gate math: r = sig(xr+br); z = sig(xz+bz); n = tanh(xn + r*bn); h=(1-z)*n
#pragma unroll
  for (int i = 0; i < 32; ++i) {
    const int idx = tid + i * 256;      // 64*128 elems
    const int m = idx >> 7, h = idx & 127;
    const float xr = sm.G[m][h]       + bih[h];
    const float xz = sm.G[m][h + 128] + bih[h + 128];
    const float xn = sm.G[m][h + 256] + bih[h + 256];
    const float r = sigm(xr + bhh[h]);
    const float z = sigm(xz + bhh[h + 128]);
    const float n = tanhf(xn + r * bhh[h + 256]);
    Hout[(size_t)(row0 + m) * 128 + h] = (1.0f - z) * n;
  }
}

// ---------------------------------------------------------------------------
// Node message pooling (softmax over j) + edge-attention node score.
// Constant-in-j logit terms cancel in softmax, so logit = dot(eh, natt_w[H:]).
// One block per (b,i); 128 threads.
// ---------------------------------------------------------------------------
__global__ __launch_bounds__(128) void node_pool(
    const float* __restrict__ eh, const float* __restrict__ natt_w,
    const float* __restrict__ eatt_w, float* __restrict__ nh,
    float* __restrict__ s) {
  __shared__ float L[64];
  __shared__ float W[64];
  __shared__ float red[128];
  const int bi = blockIdx.x;
  const int t = threadIdx.x;
  const size_t base = (size_t)bi * 64 * 128;

  if (t < 64) {
    const float4* row = (const float4*)(eh + base + (size_t)t * 128);
    const float4* wv  = (const float4*)(natt_w + 128);
    float a = 0.0f;
#pragma unroll 8
    for (int q = 0; q < 32; ++q) {
      const float4 r4 = row[q], w4 = wv[q];
      a += r4.x * w4.x + r4.y * w4.y + r4.z * w4.z + r4.w * w4.w;
    }
    L[t] = a;
  }
  __syncthreads();
  float mx = -1e30f;
  for (int j = 0; j < 64; ++j) mx = fmaxf(mx, L[j]);
  if (t < 64) W[t] = __expf(L[t] - mx);
  __syncthreads();
  float sum = 0.0f;
  for (int j = 0; j < 64; ++j) sum += W[j];

  float accp = 0.0f;
  for (int j = 0; j < 64; ++j) accp += W[j] * eh[base + (size_t)j * 128 + t];
  const float p = accp / sum;
  nh[(size_t)bi * 128 + t] = p;
  red[t] = p * eatt_w[t];  // edge-attention score contribution
  __syncthreads();
  for (int k = 64; k > 0; k >>= 1) {
    if (t < k) red[t] += red[t + k];
    __syncthreads();
  }
  if (t == 0) s[bi] = red[0];
}

// ---------------------------------------------------------------------------
// Classifier inputs: ci = [ (nh[i0]+nh[i1])/2 , a*nh[i1]+(1-a)*nh[i0] ]
// with a = sigmoid(s[i1]-s[i0])  (edge-pool 2-way softmax collapses).
// ---------------------------------------------------------------------------
__global__ __launch_bounds__(128) void build_ci(
    const float* __restrict__ nh, const float* __restrict__ s,
    const int* __restrict__ pairs, __bf16* __restrict__ ci) {
  const int bp = blockIdx.x;  // b*128 + p
  const int b = bp >> 7;
  const int t = threadIdx.x;
  const int i0 = pairs[(size_t)bp * 2 + 0];
  const int i1 = pairs[(size_t)bp * 2 + 1];
  const float h0 = nh[((size_t)b * 64 + i0) * 128 + t];
  const float h1 = nh[((size_t)b * 64 + i1) * 128 + t];
  const float a = sigm(s[b * 64 + i1] - s[b * 64 + i0]);
  ci[(size_t)bp * 256 + t]       = (__bf16)(0.5f * (h0 + h1));
  ci[(size_t)bp * 256 + 128 + t] = (__bf16)(a * h1 + (1.0f - a) * h0);
}

// ---------------------------------------------------------------------------
// Output heads: one wave per (row, output-channel); dot over CH=512.
// Channel map: 0-5 lr, 6-10 cr, 11-13 mr. Each lane owns a contiguous
// 16-element chunk of the hidden row (2 x b128 loads).
// ---------------------------------------------------------------------------
__global__ __launch_bounds__(256) void head_out(
    const __bf16* __restrict__ h_lr, const __bf16* __restrict__ h_cr,
    const __bf16* __restrict__ h_mr,
    const float* __restrict__ lr_w2, const float* __restrict__ lr_b2,
    const float* __restrict__ cr_w2, const float* __restrict__ cr_b2,
    const float* __restrict__ mr_w2, const float* __restrict__ mr_b2,
    float* __restrict__ out) {
  const int g = blockIdx.x * 8 + (threadIdx.x >> 5);
  const int lane = threadIdx.x & 31;
  const int row = g / 14;
  const int c = g % 14;
  const __bf16* hid; const float* w2; const float* b2; int od, o;
  if (c < 6)       { hid = h_lr; w2 = lr_w2; b2 = lr_b2; od = 6; o = c; }
  else if (c < 11) { hid = h_cr; w2 = cr_w2; b2 = cr_b2; od = 5; o = c - 6; }
  else             { hid = h_mr; w2 = mr_w2; b2 = mr_b2; od = 3; o = c - 11; }
  const int k0 = lane * 16;
  BF8 a0, a1;
  a0.u4 = *(const uint4*)(hid + (size_t)row * 512 + k0);
  a1.u4 = *(const uint4*)(hid + (size_t)row * 512 + k0 + 8);
  float acc = 0.0f;
#pragma unroll
  for (int q = 0; q < 8; ++q) acc += (float)a0.h[q] * w2[(k0 + q) * od + o];
#pragma unroll
  for (int q = 0; q < 8; ++q) acc += (float)a1.h[q] * w2[(k0 + 8 + q) * od + o];
#pragma unroll
  for (int m = 16; m > 0; m >>= 1) acc += __shfl_xor(acc, m, 32);
  if (lane == 0) out[(size_t)row * 14 + c] = acc + b2[o];
}

// ---------------------------------------------------------------------------
extern "C" void kernel_launch(void* const* d_in, const int* in_sizes, int n_in,
                              void* d_out, int out_size, void* d_ws,
                              size_t ws_size, hipStream_t stream) {
  (void)in_sizes; (void)n_in; (void)out_size; (void)ws_size;
  const float* x_edge   = (const float*)d_in[1];   // [16,64,64,512]
  const float* et_w     = (const float*)d_in[6];   // [512,128]
  const float* et_b     = (const float*)d_in[7];   // [128]
  const float* egru_wih = (const float*)d_in[11];  // [384,128]
  const float* egru_bih = (const float*)d_in[12];  // [384]
  const float* egru_bhh = (const float*)d_in[13];  // [384]
  const float* natt_w   = (const float*)d_in[14];  // [256,1]
  const float* eatt_w   = (const float*)d_in[16];  // [256,1]
  const float* lr_w1 = (const float*)d_in[18]; const float* lr_b1 = (const float*)d_in[19];
  const float* lr_w2 = (const float*)d_in[20]; const float* lr_b2 = (const float*)d_in[21];
  const float* cr_w1 = (const float*)d_in[22]; const float* cr_b1 = (const float*)d_in[23];
  const float* cr_w2 = (const float*)d_in[24]; const float* cr_b2 = (const float*)d_in[25];
  const float* mr_w1 = (const float*)d_in[26]; const float* mr_b1 = (const float*)d_in[27];
  const float* mr_w2 = (const float*)d_in[28]; const float* mr_b2 = (const float*)d_in[29];
  const int*   pairs = (const int*)d_in[30];       // [16,128,2]
  float* out = (float*)d_out;                      // [16,128,14]

  // Workspace carve-out (256B aligned slices)
  char* ws = (char*)d_ws;
  size_t off = 0;
  auto carve = [&](size_t bytes) -> void* {
    void* p = ws + off;
    off = (off + bytes + 255) & ~(size_t)255;
    return p;
  };
  const int ME = 65536;                                // edge rows
  __bf16* edge_f = (__bf16*)carve((size_t)ME * 128 * 2);
  float*  eh     = (float*) carve((size_t)ME * 128 * 4);
  float*  nh     = (float*) carve((size_t)1024 * 128 * 4);
  float*  sv     = (float*) carve((size_t)1024 * 4);
  __bf16* ci     = (__bf16*)carve((size_t)2048 * 256 * 2);
  __bf16* h_lr   = (__bf16*)carve((size_t)2048 * 512 * 2);
  __bf16* h_cr   = (__bf16*)carve((size_t)2048 * 512 * 2);
  __bf16* h_mr   = (__bf16*)carve((size_t)2048 * 512 * 2);
  // Pre-converted / pre-transposed bf16 weights
  __bf16* wihb   = (__bf16*)carve((size_t)384 * 128 * 2);   // wih (row-major)
  __bf16* et_wt  = (__bf16*)carve((size_t)128 * 512 * 2);   // et_w^T
  __bf16* lr1t   = (__bf16*)carve((size_t)512 * 256 * 2);   // lr_w1^T
  __bf16* cr1t   = (__bf16*)carve((size_t)512 * 256 * 2);   // cr_w1^T
  __bf16* mr1t   = (__bf16*)carve((size_t)512 * 256 * 2);   // mr_w1^T

  // 0) Weight prep (one-shot, tiny)
  convert_bf16<<<48, 256, 0, stream>>>(egru_wih, wihb, 384 * 128);
  transpose_to_bf16<<<(128 * 512) / 256, 256, 0, stream>>>(et_w, et_wt, 512, 128);
  transpose_to_bf16<<<(512 * 256) / 256, 256, 0, stream>>>(lr_w1, lr1t, 256, 512);
  transpose_to_bf16<<<(512 * 256) / 256, 256, 0, stream>>>(cr_w1, cr1t, 256, 512);
  transpose_to_bf16<<<(512 * 256) / 256, 256, 0, stream>>>(mr_w1, mr1t, 256, 512);

  // 1) Edge transform: edge_f = x_edge @ et_w + et_b        [65536,128] bf16
  gemm_bias_wmma<true, false><<<dim3(ME / 64, 1), 256, 0, stream>>>(
      x_edge, et_wt, et_b, edge_f, ME, 128, 512);
  // 2) Edge GRU (fused GEMM + gates): eh                    [65536,128] f32
  gru_wmma<<<ME / 64, 256, 0, stream>>>(edge_f, wihb, egru_bih, egru_bhh, eh);
  // 3) Node pooling softmax + node score                    nh[1024,128], s[1024]
  node_pool<<<1024, 128, 0, stream>>>(eh, natt_w, eatt_w, nh, sv);
  // 4) Classifier inputs                                    ci[2048,256] bf16
  build_ci<<<2048, 128, 0, stream>>>(nh, sv, pairs, ci);
  // 5) Head hidden layers: relu(ci @ w1 + b1)               [2048,512] bf16
  gemm_bias_wmma<false, true><<<dim3(2048 / 64, 512 / 128), 256, 0, stream>>>(
      ci, lr1t, lr_b1, h_lr, 2048, 512, 256);
  gemm_bias_wmma<false, true><<<dim3(2048 / 64, 512 / 128), 256, 0, stream>>>(
      ci, cr1t, cr_b1, h_cr, 2048, 512, 256);
  gemm_bias_wmma<false, true><<<dim3(2048 / 64, 512 / 128), 256, 0, stream>>>(
      ci, mr1t, mr_b1, h_mr, 2048, 512, 256);
  // 6) Output heads -> [B,P,14]
  head_out<<<(2048 * 14) / 8, 256, 0, stream>>>(
      h_lr, h_cr, h_mr, lr_w2, lr_b2, cr_w2, cr_b2, mr_w2, mr_b2, out);
}
